// Attention_33698313404453
// MI455X (gfx1250) — matrix operand order repaired and checked by reading, not compiled, not run on hardware
//
#include <hip/hip_runtime.h>

// ---------------------------------------------------------------------------
// Problem constants (reference: B=4, S=2048, D_IN=1024, D_H=1024, fp32)
// ---------------------------------------------------------------------------
#define BATCH 4
#define SEQ   2048
#define DIM   1024

typedef __attribute__((ext_vector_type(8)))  float  v8f;
typedef __attribute__((ext_vector_type(16))) __bf16 v16bf;

union FragBF {
    uint4 u[2];   // 32 bytes
    v16bf v;      // 16 bf16
};

// fp32 <-> bf16 (round-to-nearest-even, NaN not special-cased)
static __device__ __forceinline__ unsigned short f2bf(float f) {
    union { float f; unsigned int u; } c; c.f = f;
    unsigned int u = c.u + 0x7FFFu + ((c.u >> 16) & 1u);
    return (unsigned short)(u >> 16);
}
static __device__ __forceinline__ float bf2f(unsigned short h) {
    union { unsigned int u; float f; } c; c.u = ((unsigned int)h) << 16;
    return c.f;
}

// ---------------------------------------------------------------------------
// Streaming fp32 -> bf16 conversion (one pass; keeps conversion out of the
// GEMM hot loops which re-stage tiles many times).
// ---------------------------------------------------------------------------
__global__ __launch_bounds__(256)
void cvt_f32_bf16(const float4* __restrict__ src, uint2* __restrict__ dst, int n4)
{
    int i = blockIdx.x * 256 + threadIdx.x;
    if (i < n4) {
        float4 f = src[i];
        unsigned int lo = ((unsigned int)f2bf(f.y) << 16) | f2bf(f.x);
        unsigned int hi = ((unsigned int)f2bf(f.w) << 16) | f2bf(f.z);
        dst[i] = make_uint2(lo, hi);
    }
}

// ---------------------------------------------------------------------------
// bf16 WMMA GEMM, register-prefetch double-buffered LDS pipeline:
//   C[M,N] = alpha * A[M,K] x B      (A row-major, K contiguous, bf16)
//   B_KMAJOR=1 : B is [N,K] row-major (NT: x @ W^T, q @ k^T)
//   B_KMAJOR=0 : B is [K,N] row-major (NN: P @ V)
//   OUT_BF16   : store C as bf16 else fp32
// 256 threads (8 waves), block tile 128x128, K-step 32, ONE barrier/K-step.
// Prefetch data lives in four named uint4 registers (no arrays, no lambdas,
// no address-taking, unconditional clamped load) so it cannot be demoted to
// scratch.
// ---------------------------------------------------------------------------
template <bool B_KMAJOR, bool OUT_BF16>
__global__ __launch_bounds__(256)
void gemm_bf16_wmma(const unsigned short* __restrict__ Ap,
                    const unsigned short* __restrict__ Bp,
                    void* __restrict__ Cp, int M, int N, int K,
                    long long sA, long long sB, long long sC, float alpha)
{
    constexpr int ROWB = 80;             // 64B of K data + 16B pad per LDS row
    __shared__ __align__(16) unsigned char Alds[2][128 * ROWB];
    __shared__ __align__(16) unsigned char Blds[2][128 * ROWB];

    const int tid     = threadIdx.x;
    const int lane    = tid & 31;
    const int wave    = tid >> 5;
    const int lane_lo = lane & 15;
    const int hlf     = lane >> 4;       // half-wave select
    const int waveM   = wave >> 2;       // 0..1 -> 64-row slab
    const int waveN   = wave & 3;        // 0..3 -> 32-col slab
    const int mBase   = blockIdx.y * 128;
    const int nBase   = blockIdx.x * 128;
    const int bb      = blockIdx.z;

    const unsigned char* A  = (const unsigned char*)Ap + (size_t)bb * (size_t)sA * 2;
    const unsigned char* Bm = (const unsigned char*)Bp + (size_t)bb * (size_t)sB * 2;

    // per-thread staging coordinates: two 16B chunks for each of A and B
    const int c0 = tid, c1 = tid + 256;
    const int rA0 = c0 >> 2, cA0 = c0 & 3;          // 128 rows x 4 chunks
    const int rA1 = c1 >> 2, cA1 = c1 & 3;
    const int kB0 = c0 >> 4, nB0 = (c0 & 15) * 8;   // NN: 32 k x 16 chunks
    const int kB1 = c1 >> 4, nB1 = (c1 & 15) * 8;

    // global base pointers (kb term added as kb*64 bytes, or kb*64*N for NN B)
    const unsigned char* aG0 = A + ((size_t)(mBase + rA0) * K + cA0 * 8) * 2;
    const unsigned char* aG1 = A + ((size_t)(mBase + rA1) * K + cA1 * 8) * 2;
    const unsigned char* bG0;
    const unsigned char* bG1;
    size_t bStep;                                    // bytes per K-step for B
    if constexpr (B_KMAJOR) {
        bG0 = Bm + ((size_t)(nBase + rA0) * K + cA0 * 8) * 2;
        bG1 = Bm + ((size_t)(nBase + rA1) * K + cA1 * 8) * 2;
        bStep = 64;
    } else {
        bG0 = Bm + ((size_t)kB0 * N + nBase + nB0) * 2;
        bG1 = Bm + ((size_t)kB1 * N + nBase + nB1) * 2;
        bStep = (size_t)N * 64;
    }

    // LDS store addresses (per buffer)
    unsigned char* aL0 = &Alds[0][rA0 * ROWB + cA0 * 16];
    unsigned char* aL1 = &Alds[0][rA1 * ROWB + cA1 * 16];
    unsigned char* bL0;
    unsigned char* bL1;
    if constexpr (B_KMAJOR) {
        bL0 = &Blds[0][rA0 * ROWB + cA0 * 16];
        bL1 = &Blds[0][rA1 * ROWB + cA1 * 16];
    } else {
        bL0 = &Blds[0][nB0 * ROWB + kB0 * 2];
        bL1 = &Blds[0][nB1 * ROWB + kB1 * 2];
    }
    constexpr int BUFB = 128 * ROWB;                 // byte stride buf0 -> buf1

    v8f acc[4][2] = {};
    const int kSteps = K >> 5;

    // prime the pipeline: tile 0 in registers
    uint4 a0 = *(const uint4*)(aG0);
    uint4 a1 = *(const uint4*)(aG1);
    uint4 b0 = *(const uint4*)(bG0);
    uint4 b1 = *(const uint4*)(bG1);

    for (int kb = 0; kb < kSteps; ++kb) {
        const int cur = kb & 1;
        const int bufOff = cur * BUFB;

        // ---- ds_store current tile from registers (waits prior loads) ----
        *(uint4*)(aL0 + bufOff) = a0;
        *(uint4*)(aL1 + bufOff) = a1;
        if constexpr (B_KMAJOR) {
            *(uint4*)(bL0 + bufOff) = b0;
            *(uint4*)(bL1 + bufOff) = b1;
        } else {
            // transpose-stage [k][n] -> [n][k]: 8 bf16 per register chunk
            unsigned char* p0 = bL0 + bufOff;
            *(unsigned short*)(p0 + 0 * ROWB) = (unsigned short)(b0.x);
            *(unsigned short*)(p0 + 1 * ROWB) = (unsigned short)(b0.x >> 16);
            *(unsigned short*)(p0 + 2 * ROWB) = (unsigned short)(b0.y);
            *(unsigned short*)(p0 + 3 * ROWB) = (unsigned short)(b0.y >> 16);
            *(unsigned short*)(p0 + 4 * ROWB) = (unsigned short)(b0.z);
            *(unsigned short*)(p0 + 5 * ROWB) = (unsigned short)(b0.z >> 16);
            *(unsigned short*)(p0 + 6 * ROWB) = (unsigned short)(b0.w);
            *(unsigned short*)(p0 + 7 * ROWB) = (unsigned short)(b0.w >> 16);
            unsigned char* p1 = bL1 + bufOff;
            *(unsigned short*)(p1 + 0 * ROWB) = (unsigned short)(b1.x);
            *(unsigned short*)(p1 + 1 * ROWB) = (unsigned short)(b1.x >> 16);
            *(unsigned short*)(p1 + 2 * ROWB) = (unsigned short)(b1.y);
            *(unsigned short*)(p1 + 3 * ROWB) = (unsigned short)(b1.y >> 16);
            *(unsigned short*)(p1 + 4 * ROWB) = (unsigned short)(b1.z);
            *(unsigned short*)(p1 + 5 * ROWB) = (unsigned short)(b1.z >> 16);
            *(unsigned short*)(p1 + 6 * ROWB) = (unsigned short)(b1.w);
            *(unsigned short*)(p1 + 7 * ROWB) = (unsigned short)(b1.w >> 16);
        }

        // ---- unconditionally issue next tile (clamped; no wait here) ----
        const int kn = (kb + 1 < kSteps) ? (kb + 1) : (kSteps - 1);
        const size_t aOff = (size_t)kn * 64;
        const size_t bOff = (size_t)kn * bStep;
        a0 = *(const uint4*)(aG0 + aOff);
        a1 = *(const uint4*)(aG1 + aOff);
        b0 = *(const uint4*)(bG0 + bOff);
        b1 = *(const uint4*)(bG1 + bOff);

        __syncthreads();                  // single barrier per K-step

        // ---- fragments per documented 16-bit VGPR layouts ----
        FragBF aF[4], bF[2];
        #pragma unroll
        for (int ms = 0; ms < 4; ++ms) {
            // A 16x32: lanes<16 K{0..7,16..23}, lanes>=16 K{8..15,24..31}
            const unsigned char* p = &Alds[0][0] + bufOff + (waveM * 64 + ms * 16 + lane_lo) * ROWB;
            aF[ms].u[0] = *(const uint4*)(p + hlf * 16);
            aF[ms].u[1] = *(const uint4*)(p + 32 + hlf * 16);
        }
        #pragma unroll
        for (int ns = 0; ns < 2; ++ns) {
            // B 32x16: lanes<16 K{0..7,16..23}, lanes>=16 K{8..15,24..31}
            const unsigned char* p = &Blds[0][0] + bufOff + (waveN * 32 + ns * 16 + lane_lo) * ROWB;
            bF[ns].u[0] = *(const uint4*)(p + hlf * 16);
            bF[ns].u[1] = *(const uint4*)(p + 32 + hlf * 16);
        }
        #pragma unroll
        for (int ms = 0; ms < 4; ++ms)
            #pragma unroll
            for (int ns = 0; ns < 2; ++ns)
                acc[ms][ns] = __builtin_amdgcn_wmma_f32_16x16x32_bf16(
                    false, aF[ms].v, false, bF[ns].v,
                    (short)0, acc[ms][ns], false, false);
    }

    // epilogue: C 16x16 f32 layout: VGPR j -> M = j + 8*half, N = lane_lo
    unsigned char* C = (unsigned char*)Cp + (size_t)bb * (size_t)sC * (OUT_BF16 ? 2 : 4);
    #pragma unroll
    for (int ms = 0; ms < 4; ++ms)
        #pragma unroll
        for (int ns = 0; ns < 2; ++ns)
            #pragma unroll
            for (int j = 0; j < 8; ++j) {
                int row = mBase + waveM * 64 + ms * 16 + hlf * 8 + j;
                int col = nBase + waveN * 32 + ns * 16 + lane_lo;
                float v = acc[ms][ns][j] * alpha;
                if constexpr (OUT_BF16)
                    *(unsigned short*)(C + ((size_t)row * N + col) * 2) = f2bf(v);
                else
                    *(float*)(C + ((size_t)row * N + col) * 4) = v;
            }
}

// ---------------------------------------------------------------------------
// Column-softmax over the QUERY axis (reference softmax dim=1).
// Mask substitutes the VALUE 1e-9 (entry still participates in softmax).
// ---------------------------------------------------------------------------
__global__ __launch_bounds__(256)
void colstats_kernel(const unsigned short* __restrict__ score,
                     const unsigned char* __restrict__ mask,
                     float* __restrict__ cmax, float* __restrict__ csum)
{
    int idx = blockIdx.x * 256 + threadIdx.x;        // over BATCH*SEQ columns
    int b = idx >> 11;                               // / SEQ
    int k = idx & (SEQ - 1);
    size_t base = (size_t)b * SEQ * SEQ + k;
    float m = -3.0e38f, s = 0.0f;
    for (int q = 0; q < SEQ; ++q) {
        size_t o = base + (size_t)q * SEQ;
        float v = mask[o] ? 1e-9f : bf2f(score[o]);
        float nm = fmaxf(m, v);
        s = s * __expf(m - nm) + __expf(v - nm);
        m = nm;
    }
    cmax[idx] = m;
    csum[idx] = s;
}

__global__ __launch_bounds__(256)
void colnorm_kernel(unsigned short* __restrict__ score,
                    const unsigned char* __restrict__ mask,
                    const float* __restrict__ cmax, const float* __restrict__ csum)
{
    size_t idx = (size_t)blockIdx.x * 256 + threadIdx.x;   // over BATCH*SEQ*SEQ
    int b = (int)(idx >> 22);                              // / (SEQ*SEQ)
    int k = (int)(idx & (SEQ - 1));
    float v = mask[idx] ? 1e-9f : bf2f(score[idx]);
    int si = b * SEQ + k;
    float p = __expf(v - cmax[si]) / csum[si];
    score[idx] = f2bf(p);
}

// ---------------------------------------------------------------------------
// Launch: convert -> projections -> scores -> column softmax -> PV
// ---------------------------------------------------------------------------
extern "C" void kernel_launch(void* const* d_in, const int* in_sizes, int n_in,
                              void* d_out, int out_size, void* d_ws, size_t ws_size,
                              hipStream_t stream)
{
    (void)in_sizes; (void)n_in; (void)out_size; (void)ws_size;
    const float*         seq1 = (const float*)d_in[0];
    const float*         seq2 = (const float*)d_in[1];
    const unsigned char* mask = (const unsigned char*)d_in[2];  // jnp.bool_ = 1 byte
    const float*         Wq   = (const float*)d_in[3];
    const float*         Wk   = (const float*)d_in[4];
    const float*         Wv   = (const float*)d_in[5];

    // workspace layout (bytes)
    const size_t seqBytes   = (size_t)BATCH * SEQ * DIM * 2;   // 16 MiB
    const size_t wBytes     = (size_t)DIM * DIM * 2;           //  2 MiB
    const size_t scoreBytes = (size_t)BATCH * SEQ * SEQ * 2;   // 32 MiB
    unsigned char* ws = (unsigned char*)d_ws;
    unsigned short* seq1b = (unsigned short*)(ws);
    unsigned short* seq2b = (unsigned short*)(ws + seqBytes);
    unsigned short* Wqb   = (unsigned short*)(ws + 2 * seqBytes);
    unsigned short* Wkb   = (unsigned short*)(ws + 2 * seqBytes + wBytes);
    unsigned short* Wvb   = (unsigned short*)(ws + 2 * seqBytes + 2 * wBytes);
    unsigned short* qb    = (unsigned short*)(ws + 2 * seqBytes + 3 * wBytes);
    unsigned short* kb    = (unsigned short*)(ws + 3 * seqBytes + 3 * wBytes);
    unsigned short* vb    = (unsigned short*)(ws + 4 * seqBytes + 3 * wBytes);
    unsigned short* sc    = (unsigned short*)(ws + 5 * seqBytes + 3 * wBytes);
    float* cmax = (float*)(ws + 5 * seqBytes + 3 * wBytes + scoreBytes);
    float* csum = cmax + (size_t)BATCH * SEQ;

    dim3 blk(256);

    // 0) one-time fp32 -> bf16 conversions
    const int seq4 = (BATCH * SEQ * DIM) / 4;     // 2,097,152
    const int w4   = (DIM * DIM) / 4;             //   262,144
    cvt_f32_bf16<<<seq4 / 256, blk, 0, stream>>>((const float4*)seq1, (uint2*)seq1b, seq4);
    cvt_f32_bf16<<<seq4 / 256, blk, 0, stream>>>((const float4*)seq2, (uint2*)seq2b, seq4);
    cvt_f32_bf16<<<w4 / 256,   blk, 0, stream>>>((const float4*)Wq,   (uint2*)Wqb,   w4);
    cvt_f32_bf16<<<w4 / 256,   blk, 0, stream>>>((const float4*)Wk,   (uint2*)Wkb,   w4);
    cvt_f32_bf16<<<w4 / 256,   blk, 0, stream>>>((const float4*)Wv,   (uint2*)Wvb,   w4);

    // 1..3) projections: [8192,1024] x [1024,1024]^T -> bf16
    dim3 gProj(DIM / 128, (BATCH * SEQ) / 128, 1);
    gemm_bf16_wmma<true, true><<<gProj, blk, 0, stream>>>(
        seq1b, Wqb, qb, BATCH * SEQ, DIM, DIM, 0, 0, 0, 1.0f);
    gemm_bf16_wmma<true, true><<<gProj, blk, 0, stream>>>(
        seq2b, Wkb, kb, BATCH * SEQ, DIM, DIM, 0, 0, 0, 1.0f);
    gemm_bf16_wmma<true, true><<<gProj, blk, 0, stream>>>(
        seq2b, Wvb, vb, BATCH * SEQ, DIM, DIM, 0, 0, 0, 1.0f);

    // 4) scores = (q @ k^T) / sqrt(DIM), batched, bf16
    dim3 gScore(SEQ / 128, SEQ / 128, BATCH);
    gemm_bf16_wmma<true, true><<<gScore, blk, 0, stream>>>(
        qb, kb, sc, SEQ, SEQ, DIM,
        (long long)SEQ * DIM, (long long)SEQ * DIM, (long long)SEQ * SEQ,
        1.0f / 32.0f);

    // 5) column softmax (over query axis), mask -> value 1e-9
    colstats_kernel<<<(BATCH * SEQ) / 256, blk, 0, stream>>>(sc, mask, cmax, csum);
    colnorm_kernel<<<(unsigned)(((size_t)BATCH * SEQ * SEQ) / 256), blk, 0, stream>>>(
        sc, mask, cmax, csum);

    // 6) out = P @ V  (NN gemm), fp32 output
    dim3 gPV(DIM / 128, SEQ / 128, BATCH);
    gemm_bf16_wmma<false, false><<<gPV, blk, 0, stream>>>(
        sc, vb, d_out, SEQ, DIM, SEQ,
        (long long)SEQ * SEQ, (long long)SEQ * DIM, (long long)SEQ * DIM,
        1.0f);
}
